// GATNet_54090818126587
// MI455X (gfx1250) — matrix operand order repaired and checked by reading, not compiled, not run on hardware
//
#include <hip/hip_runtime.h>
#include <hip/hip_bf16.h>

#define F1 512
#define H1DIM 64
#define HEADS1 8
#define OUT1 8
#define NC 40

typedef __bf16 bf16;
typedef __attribute__((ext_vector_type(16))) bf16 v16bf;
typedef __attribute__((ext_vector_type(8)))  float v8f;

// ---------- helpers ----------
__device__ __forceinline__ bf16 f2bf(float f) {
  unsigned u = __float_as_uint(f);
  unsigned r = (u + 0x7FFFu + ((u >> 16) & 1u)) >> 16;  // RNE
  unsigned short s = (unsigned short)r;
  return __builtin_bit_cast(bf16, s);
}
__device__ __forceinline__ float bf2f(bf16 h) {
  unsigned short s = __builtin_bit_cast(unsigned short, h);
  return __uint_as_float(((unsigned)s) << 16);
}
__device__ __forceinline__ float leaky02(float v) { return v > 0.f ? v : 0.2f * v; }
__device__ __forceinline__ void atomicMaxF(float* a, float v) {
  if (v >= 0.f) atomicMax((int*)a, __float_as_int(v));
  else          atomicMin((unsigned int*)a, __float_as_uint(v));
}

// ---------- setup kernels ----------
__global__ void gat_fill_kernel(float* p, float v, int n) {
  int t = blockIdx.x * blockDim.x + threadIdx.x;
  if (t < n) p[t] = v;
}

// Pack W[K,Ncols] (row-major) into per-lane WMMA B layout, split bf16 hi/lo.
// idx = (((ks*ctiles)+ct)*32 + lane)*16 + j ; element j: k = 8*half + j (j<8), 16 + 8*half + (j-8) (j>=8).
__global__ void gat_pack_w_kernel(const float* __restrict__ W, bf16* __restrict__ Whi,
                                  bf16* __restrict__ Wlo, int ksteps, int ctiles,
                                  int Kdim, int Ncols) {
  int idx = blockIdx.x * blockDim.x + threadIdx.x;
  int total = ksteps * ctiles * 32 * 16;
  if (idx >= total) return;
  int j    = idx & 15;
  int lane = (idx >> 4) & 31;
  int rest = idx >> 9;
  int ct   = rest % ctiles;
  int ks   = rest / ctiles;
  int k    = ks * 32 + ((j >> 3) << 4) + ((lane >> 4) << 3) + (j & 7);   // B: interleaved K groups of 8, matching A
  int col  = ct * 16 + (lane & 15);
  float v = 0.f;
  if (k < Kdim && col < Ncols) v = W[(size_t)k * Ncols + col];
  bf16 hi = f2bf(v);
  Whi[idx] = hi;
  Wlo[idx] = f2bf(v - bf2f(hi));
}

// ---------- GEMM1: h1 = x @ W1  (bf16 hi/lo 3-product WMMA) ----------
__global__ void gat_gemm1_wmma(const float* __restrict__ x, const bf16* __restrict__ Whi,
                               const bf16* __restrict__ Wlo, float* __restrict__ h1, int N) {
  int wave = threadIdx.x >> 5;
  int lane = threadIdx.x & 31;
  int rowBase = blockIdx.x * 128 + wave * 16;
  int m = lane & 15;
  int kgrp = (lane >> 4) << 3;                  // A interleave: 0 or 8
  int row = rowBase + m; if (row >= N) row = N - 1;
  const float* xr = x + (size_t)row * F1;

  v8f acc[4] = {v8f{}, v8f{}, v8f{}, v8f{}};
  for (int ks = 0; ks < F1 / 32; ++ks) {
    int k0 = ks * 32;
    v16bf ahi, alo;
#pragma unroll
    for (int j = 0; j < 8; ++j) {
      float f0 = xr[k0 + kgrp + j];
      float f1 = xr[k0 + 16 + kgrp + j];
      bf16 h0 = f2bf(f0), h1e = f2bf(f1);
      ahi[j] = h0;  ahi[8 + j] = h1e;
      alo[j] = f2bf(f0 - bf2f(h0));
      alo[8 + j] = f2bf(f1 - bf2f(h1e));
    }
#pragma unroll
    for (int t = 0; t < 4; ++t) {
      const v16bf bhi = *(const v16bf*)(Whi + ((size_t)(ks * 4 + t) * 32 + lane) * 16);
      const v16bf blo = *(const v16bf*)(Wlo + ((size_t)(ks * 4 + t) * 32 + lane) * 16);
      acc[t] = __builtin_amdgcn_wmma_f32_16x16x32_bf16(false, ahi, false, bhi, (short)0, acc[t], false, false);
      acc[t] = __builtin_amdgcn_wmma_f32_16x16x32_bf16(false, ahi, false, blo, (short)0, acc[t], false, false);
      acc[t] = __builtin_amdgcn_wmma_f32_16x16x32_bf16(false, alo, false, bhi, (short)0, acc[t], false, false);
    }
  }
  int n = lane & 15;
  int mBase = rowBase + ((lane >> 4) << 3);
#pragma unroll
  for (int t = 0; t < 4; ++t)
#pragma unroll
    for (int r = 0; r < 8; ++r) {
      int rr = mBase + r;
      if (rr < N) h1[(size_t)rr * H1DIM + t * 16 + n] = acc[t][r];
    }
}

// ---------- GEMM2: h2 = ELU(out1 + b1) @ W2 (K=64, Npad=48) ----------
__global__ void gat_gemm2_wmma(const float* __restrict__ out1, const float* __restrict__ b1,
                               const bf16* __restrict__ Whi, const bf16* __restrict__ Wlo,
                               float* __restrict__ h2, int N) {
  int wave = threadIdx.x >> 5;
  int lane = threadIdx.x & 31;
  int rowBase = blockIdx.x * 128 + wave * 16;
  int m = lane & 15;
  int kgrp = (lane >> 4) << 3;
  int row = rowBase + m; if (row >= N) row = N - 1;
  const float* ar = out1 + (size_t)row * H1DIM;

  v8f acc[3] = {v8f{}, v8f{}, v8f{}};
  for (int ks = 0; ks < 2; ++ks) {
    int k0 = ks * 32;
    v16bf ahi, alo;
#pragma unroll
    for (int j = 0; j < 8; ++j) {
      int ka = k0 + kgrp + j, kb = k0 + 16 + kgrp + j;
      float f0 = ar[ka] + b1[ka];  f0 = f0 > 0.f ? f0 : (expf(f0) - 1.0f);
      float f1 = ar[kb] + b1[kb];  f1 = f1 > 0.f ? f1 : (expf(f1) - 1.0f);
      bf16 h0 = f2bf(f0), h1e = f2bf(f1);
      ahi[j] = h0;  ahi[8 + j] = h1e;
      alo[j] = f2bf(f0 - bf2f(h0));
      alo[8 + j] = f2bf(f1 - bf2f(h1e));
    }
#pragma unroll
    for (int t = 0; t < 3; ++t) {
      const v16bf bhi = *(const v16bf*)(Whi + ((size_t)(ks * 3 + t) * 32 + lane) * 16);
      const v16bf blo = *(const v16bf*)(Wlo + ((size_t)(ks * 3 + t) * 32 + lane) * 16);
      acc[t] = __builtin_amdgcn_wmma_f32_16x16x32_bf16(false, ahi, false, bhi, (short)0, acc[t], false, false);
      acc[t] = __builtin_amdgcn_wmma_f32_16x16x32_bf16(false, ahi, false, blo, (short)0, acc[t], false, false);
      acc[t] = __builtin_amdgcn_wmma_f32_16x16x32_bf16(false, alo, false, bhi, (short)0, acc[t], false, false);
    }
  }
  int n = lane & 15;
  int mBase = rowBase + ((lane >> 4) << 3);
#pragma unroll
  for (int t = 0; t < 3; ++t)
#pragma unroll
    for (int r = 0; r < 8; ++r) {
      int rr = mBase + r;
      int col = t * 16 + n;
      if (rr < N && col < NC) h2[(size_t)rr * NC + col] = acc[t][r];
    }
}

// ---------- attention coefficient tables ----------
__global__ void gat_att1_kernel(const float* __restrict__ h1, const float* __restrict__ asv,
                                const float* __restrict__ adv, float* __restrict__ aS,
                                float* __restrict__ aD, int N) {
  int t = blockIdx.x * blockDim.x + threadIdx.x;
  if (t >= N * HEADS1) return;
  int n = t >> 3, h = t & 7;
  const float* hp = h1 + (size_t)n * H1DIM + h * OUT1;
  float s = 0.f, d = 0.f;
#pragma unroll
  for (int c = 0; c < OUT1; ++c) { float v = hp[c]; s += v * asv[h * OUT1 + c]; d += v * adv[h * OUT1 + c]; }
  aS[t] = s; aD[t] = d;
}

__global__ void gat_att2_kernel(const float* __restrict__ h2, const float* __restrict__ asv,
                                const float* __restrict__ adv, float* __restrict__ aS,
                                float* __restrict__ aD, int N) {
  int n = blockIdx.x * blockDim.x + threadIdx.x;
  if (n >= N) return;
  const float* hp = h2 + (size_t)n * NC;
  float s = 0.f, d = 0.f;
  for (int c = 0; c < NC; ++c) { float v = hp[c]; s += v * asv[c]; d += v * adv[c]; }
  aS[n] = s; aD[n] = d;
}

// ---------- edge passes (segment softmax over dst) ----------
__global__ void gat_edge_max_kernel(const int* __restrict__ src, const int* __restrict__ dst,
                                    const float* __restrict__ aS, const float* __restrict__ aD,
                                    float* __restrict__ m, int total, int hshift) {
  int t = blockIdx.x * blockDim.x + threadIdx.x;
  if (t >= total) return;
  int e = t >> hshift;
  int h = t & ((1 << hshift) - 1);
  int H = 1 << hshift;
  int s = src[e], d = dst[e];
  float v = leaky02(aS[s * H + h] + aD[d * H + h]);
  atomicMaxF(&m[d * H + h], v);
}

__global__ void gat_edge_sum_kernel(const int* __restrict__ src, const int* __restrict__ dst,
                                    const float* __restrict__ aS, const float* __restrict__ aD,
                                    const float* __restrict__ m, float* __restrict__ den,
                                    int total, int hshift) {
  int t = blockIdx.x * blockDim.x + threadIdx.x;
  if (t >= total) return;
  int e = t >> hshift;
  int h = t & ((1 << hshift) - 1);
  int H = 1 << hshift;
  int s = src[e], d = dst[e];
  float v = leaky02(aS[s * H + h] + aD[d * H + h]);
  atomicAdd(&den[d * H + h], expf(v - m[d * H + h]));
}

// One thread per (edge, head): alpha computed once, head slice gathered as 2x float4,
// 8 scalar f32 atomic adds into the L2-resident out1 buffer.
__global__ void gat_edge_agg1_kernel(const int* __restrict__ src, const int* __restrict__ dst,
                                     const float* __restrict__ aS, const float* __restrict__ aD,
                                     const float* __restrict__ m, const float* __restrict__ den,
                                     const float* __restrict__ h1, float* __restrict__ out1, int E) {
  int t = blockIdx.x * blockDim.x + threadIdx.x;
  if (t >= E * HEADS1) return;
  int e = t >> 3, h = t & 7;
  int s = src[e], d = dst[e];
  float v = leaky02(aS[s * 8 + h] + aD[d * 8 + h]);
  float alpha = expf(v - m[d * 8 + h]) / (den[d * 8 + h] + 1e-16f);
  const float4* hp = (const float4*)(h1 + (size_t)s * H1DIM + h * OUT1);
  float4 v0 = hp[0], v1 = hp[1];
  float* op = out1 + (size_t)d * H1DIM + h * OUT1;
  atomicAdd(op + 0, alpha * v0.x);
  atomicAdd(op + 1, alpha * v0.y);
  atomicAdd(op + 2, alpha * v0.z);
  atomicAdd(op + 3, alpha * v0.w);
  atomicAdd(op + 4, alpha * v1.x);
  atomicAdd(op + 5, alpha * v1.y);
  atomicAdd(op + 6, alpha * v1.z);
  atomicAdd(op + 7, alpha * v1.w);
}

// One thread per (edge, 4-channel group): 10 groups cover NC=40. float4 gather stays
// 16B-aligned (row stride 160B).
__global__ void gat_edge_agg2_kernel(const int* __restrict__ src, const int* __restrict__ dst,
                                     const float* __restrict__ aS, const float* __restrict__ aD,
                                     const float* __restrict__ m, const float* __restrict__ den,
                                     const float* __restrict__ h2, float* __restrict__ out2, int E) {
  int t = blockIdx.x * blockDim.x + threadIdx.x;
  if (t >= E * 10) return;
  int e = t / 10;
  int g = t - e * 10;
  int s = src[e], d = dst[e];
  float v = leaky02(aS[s] + aD[d]);
  float alpha = expf(v - m[d]) / (den[d] + 1e-16f);
  float4 hv = *(const float4*)(h2 + (size_t)s * NC + g * 4);
  float* op = out2 + (size_t)d * NC + g * 4;
  atomicAdd(op + 0, alpha * hv.x);
  atomicAdd(op + 1, alpha * hv.y);
  atomicAdd(op + 2, alpha * hv.z);
  atomicAdd(op + 3, alpha * hv.w);
}

// ---------- final log_softmax ----------
__global__ void gat_logsoftmax_kernel(const float* __restrict__ out2, const float* __restrict__ b2,
                                      float* __restrict__ out, int N) {
  int n = blockIdx.x * blockDim.x + threadIdx.x;
  if (n >= N) return;
  const float* r = out2 + (size_t)n * NC;
  float mx = -3.4e38f;
  for (int c = 0; c < NC; ++c) mx = fmaxf(mx, r[c] + b2[c]);
  float sum = 0.f;
  for (int c = 0; c < NC; ++c) sum += expf(r[c] + b2[c] - mx);
  float ls = logf(sum);
  for (int c = 0; c < NC; ++c) out[(size_t)n * NC + c] = r[c] + b2[c] - mx - ls;
}

// ---------- host launcher ----------
extern "C" void kernel_launch(void* const* d_in, const int* in_sizes, int n_in,
                              void* d_out, int out_size, void* d_ws, size_t ws_size,
                              hipStream_t stream) {
  const float* x   = (const float*)d_in[0];
  const int*   ei  = (const int*)d_in[1];
  const float* W1  = (const float*)d_in[2];
  const float* as1 = (const float*)d_in[3];
  const float* ad1 = (const float*)d_in[4];
  const float* b1  = (const float*)d_in[5];
  const float* W2  = (const float*)d_in[6];
  const float* as2 = (const float*)d_in[7];
  const float* ad2 = (const float*)d_in[8];
  const float* b2  = (const float*)d_in[9];
  float* out = (float*)d_out;

  int N = in_sizes[0] / F1;
  int E = in_sizes[1] / 2;
  const int* src = ei;
  const int* dst = ei + E;

  float* w    = (float*)d_ws;
  float* h1   = w;                            // N*64 (later aliased as h2: N*40)
  float* aS1  = h1   + (size_t)N * 64;        // N*8
  float* aD1  = aS1  + (size_t)N * 8;         // N*8
  float* m1   = aD1  + (size_t)N * 8;         // N*8
  float* den1 = m1   + (size_t)N * 8;         // N*8
  float* out1 = den1 + (size_t)N * 8;         // N*64
  float* out2 = out1 + (size_t)N * 64;        // N*40
  float* aS2  = out2 + (size_t)N * NC;        // N
  float* aD2  = aS2  + N;                     // N
  float* m2   = aD2  + N;                     // N
  float* den2 = m2   + N;                     // N
  bf16*  W1hi = (bf16*)(den2 + N);            // 16*4*32*16 = 32768
  bf16*  W1lo = W1hi + 32768;
  bf16*  W2hi = W1lo + 32768;                 // 2*3*32*16 = 3072
  bf16*  W2lo = W2hi + 3072;
  float* h2   = h1;                           // alias (h1 dead after layer-1 aggregation)

  // init accumulators (every call; all on stream, graph-capturable)
  hipMemsetAsync(den1, 0, (size_t)N * 8 * sizeof(float), stream);
  hipMemsetAsync(out1, 0, (size_t)N * 64 * sizeof(float), stream);
  hipMemsetAsync(den2, 0, (size_t)N * sizeof(float), stream);
  hipMemsetAsync(out2, 0, (size_t)N * NC * sizeof(float), stream);
  gat_fill_kernel<<<(N * 8 + 255) / 256, 256, 0, stream>>>(m1, -__builtin_inff(), N * 8);
  gat_fill_kernel<<<(N + 255) / 256, 256, 0, stream>>>(m2, -__builtin_inff(), N);
  gat_pack_w_kernel<<<(32768 + 255) / 256, 256, 0, stream>>>(W1, W1hi, W1lo, 16, 4, F1, H1DIM);
  gat_pack_w_kernel<<<(3072 + 255) / 256, 256, 0, stream>>>(W2, W2hi, W2lo, 2, 3, H1DIM, NC);

  int gb = (N + 127) / 128;

  // ---- layer 1 ----
  gat_gemm1_wmma<<<gb, 256, 0, stream>>>(x, W1hi, W1lo, h1, N);
  gat_att1_kernel<<<(N * 8 + 255) / 256, 256, 0, stream>>>(h1, as1, ad1, aS1, aD1, N);
  {
    int t1 = E * 8;
    gat_edge_max_kernel<<<(t1 + 255) / 256, 256, 0, stream>>>(src, dst, aS1, aD1, m1, t1, 3);
    gat_edge_sum_kernel<<<(t1 + 255) / 256, 256, 0, stream>>>(src, dst, aS1, aD1, m1, den1, t1, 3);
    gat_edge_agg1_kernel<<<(t1 + 255) / 256, 256, 0, stream>>>(src, dst, aS1, aD1, m1, den1, h1, out1, E);
  }

  // ---- layer 2 (ELU + b1 fused into GEMM2 A-load) ----
  gat_gemm2_wmma<<<gb, 256, 0, stream>>>(out1, b1, W2hi, W2lo, h2, N);
  gat_att2_kernel<<<(N + 255) / 256, 256, 0, stream>>>(h2, as2, ad2, aS2, aD2, N);
  {
    gat_edge_max_kernel<<<(E + 255) / 256, 256, 0, stream>>>(src, dst, aS2, aD2, m2, E, 0);
    gat_edge_sum_kernel<<<(E + 255) / 256, 256, 0, stream>>>(src, dst, aS2, aD2, m2, den2, E, 0);
    int t2 = E * 10;
    gat_edge_agg2_kernel<<<(t2 + 255) / 256, 256, 0, stream>>>(src, dst, aS2, aD2, m2, den2, h2, out2, E);
  }

  gat_logsoftmax_kernel<<<(N + 255) / 256, 256, 0, stream>>>(out2, b2, out, N);
}
